// ConvModule_27479200759862
// MI455X (gfx1250) — matrix operand order repaired and checked by reading, not compiled, not run on hardware
//
#include <hip/hip_runtime.h>
#include <hip/hip_bf16.h>

// ---------------------------------------------------------------------------
// Conformer conv-module, MI455X (gfx1250), bf16 WMMA pipeline.
//   B=16, S=2048, C=512, K=5.  ~481 GFLOP of GEMM work, <1GB traffic
//   -> compute bound -> v_wmma_f32_16x16x32_bf16 everywhere.
//   Conv kernels: uniform fast path (no boundary predication) for interior
//   N-tiles; edge path only for blockIdx.y == 0 / last.
// ---------------------------------------------------------------------------

#define BB 16
#define SS 2048
#define CC 512
#define C2 1024
#define KW 5

typedef __bf16 bf16_t;
typedef bf16_t v16bf __attribute__((ext_vector_type(16)));
typedef float  v8f   __attribute__((ext_vector_type(8)));

union BFrag { v16bf v; uint4 u[2]; };

// float -> bf16 round-to-nearest-even (bit trick)
__device__ __forceinline__ unsigned short f2bf(float f) {
  union { float f; unsigned u; } v; v.f = f;
  unsigned r = v.u + 0x7FFFu + ((v.u >> 16) & 1u);
  return (unsigned short)(r >> 16);
}

__device__ __forceinline__ float sigmoidf(float x) {
  return 1.0f / (1.0f + __expf(-x));
}

// A fragment: 16x32 bf16, row-major A with leading dim ldk.
__device__ __forceinline__ v16bf load_a(const unsigned short* __restrict__ A0,
                                        int ldk, int kb, int lane) {
  const unsigned short* p = A0 + (size_t)(lane & 15) * ldk + kb + ((lane >> 4) << 3);
  BFrag r;
  r.u[0] = *(const uint4*)(p);
  r.u[1] = *(const uint4*)(p + 16);
  return r.v;
}

// B fragment: 32x16 bf16 from X[s][k] (k contiguous). Unconditional.
__device__ __forceinline__ v16bf load_b_fast(const unsigned short* __restrict__ X0,
                                             int ldk, int srow, int kb, int lane) {
  const unsigned short* p = X0 + (size_t)srow * ldk + kb + ((lane >> 4) << 3);
  BFrag r;
  r.u[0] = *(const uint4*)(p);
  r.u[1] = *(const uint4*)(p + 16);
  return r.v;
}

// Predicated variant (conv padding) for edge tiles only.
__device__ __forceinline__ v16bf load_b_edge(const unsigned short* __restrict__ X0,
                                             int ldk, int srow, int smax, int kb, int lane) {
  BFrag r;
  if ((unsigned)srow < (unsigned)smax) {
    const unsigned short* p = X0 + (size_t)srow * ldk + kb + ((lane >> 4) << 3);
    r.u[0] = *(const uint4*)(p);
    r.u[1] = *(const uint4*)(p + 16);
  } else {
    r.u[0] = make_uint4(0u, 0u, 0u, 0u);
    r.u[1] = make_uint4(0u, 0u, 0u, 0u);
  }
  return r.v;
}

template <bool EDGE>
__device__ __forceinline__ v16bf load_b(const unsigned short* __restrict__ X0,
                                        int ldk, int srow, int smax, int kb, int lane) {
  if (EDGE) return load_b_edge(X0, ldk, srow, smax, kb, lane);
  return load_b_fast(X0, ldk, srow, kb, lane);
}

__device__ __forceinline__ uint4 pack8bf(const float* v) {
  uint4 r;
  r.x = (unsigned)f2bf(v[0]) | ((unsigned)f2bf(v[1]) << 16);
  r.y = (unsigned)f2bf(v[2]) | ((unsigned)f2bf(v[3]) << 16);
  r.z = (unsigned)f2bf(v[4]) | ((unsigned)f2bf(v[5]) << 16);
  r.w = (unsigned)f2bf(v[6]) | ((unsigned)f2bf(v[7]) << 16);
  return r;
}

// ---------------------------------------------------------------------------
// prep: cast x -> bf16, repack weights per-tap row-major bf16, zero BN stats
// ---------------------------------------------------------------------------
__global__ void k_prep(const float* __restrict__ x, const float* __restrict__ wd,
                       const float* __restrict__ w1, const float* __restrict__ w2,
                       unsigned short* __restrict__ xb, unsigned short* __restrict__ wdp,
                       unsigned short* __restrict__ w1p, unsigned short* __restrict__ w2p,
                       float* __restrict__ st) {
  const size_t NX = (size_t)BB * SS * CC;
  const size_t ND = (size_t)CC * CC * KW;
  const size_t N1 = (size_t)C2 * CC;
  const size_t N2 = (size_t)C2 * C2 * KW;
  size_t i = (size_t)blockIdx.x * blockDim.x + threadIdx.x;
  size_t stride = (size_t)gridDim.x * blockDim.x;
  if (i < 1024) st[i] = 0.0f;  // zero BN sum / sumsq accumulators
  for (size_t t = i; t < NX + ND + N1 + N2; t += stride) {
    if (t < NX) {
      xb[t] = f2bf(x[t]);
    } else if (t < NX + ND) {
      size_t j = t - NX;                 // (co, ci, k) row-major
      size_t co = j / (CC * KW);
      size_t rr = j % (CC * KW);
      size_t ci = rr / KW, k = rr % KW;
      wdp[((k * CC) + co) * CC + ci] = f2bf(wd[j]);
    } else if (t < NX + ND + N1) {
      size_t j = t - NX - ND;            // already row-major (o, c)
      w1p[j] = f2bf(w1[j]);
    } else {
      size_t j = t - NX - ND - N1;       // (o, i, k)
      size_t co = j / (C2 * KW);
      size_t rr = j % (C2 * KW);
      size_t ci = rr / KW, k = rr % KW;
      w2p[((k * C2) + co) * C2 + ci] = f2bf(w2[j]);
    }
  }
}

// ---------------------------------------------------------------------------
// conv1 main loop (templated on EDGE)
// ---------------------------------------------------------------------------
template <bool EDGE>
__device__ __forceinline__ void conv1_loop(const unsigned short* __restrict__ X,
                                           const unsigned short* __restrict__ wdp,
                                           int m0, int s0, int lane, v8f acc[4]) {
  const int nl = lane & 15;
  for (int t = 0; t < KW; ++t) {
    const unsigned short* A = wdp + ((size_t)t * CC + m0) * CC;
    for (int k = 0; k < CC; k += 32) {
      v16bf af = load_a(A, CC, k, lane);
#pragma unroll
      for (int j = 0; j < 4; ++j) {
        int s = s0 + j * 16 + nl + t - 2;
        v16bf bf = load_b<EDGE>(X, CC, s, SS, k, lane);
        acc[j] = __builtin_amdgcn_wmma_f32_16x16x32_bf16(
            false, af, false, bf, (short)0, acc[j], false, false);
      }
    }
  }
}

// conv1: y[b,s,co] = sum_{t,ci} x[b, s+t-2, ci] * w[co,ci,t] + dcnn_b[co]
// grid (C/128, S/64, B), 256 thr (8 waves), wave tile 16(M) x 64(N)
__global__ __launch_bounds__(256)
void k_conv1(const unsigned short* __restrict__ xb, const unsigned short* __restrict__ wdp,
             const float* __restrict__ bias, float* __restrict__ y) {
  const int lane = threadIdx.x & 31, wv = threadIdx.x >> 5;
  const int nl = lane & 15, hf = lane >> 4;
  const int b  = blockIdx.z;
  const int m0 = blockIdx.x * 128 + wv * 16;
  const int s0 = blockIdx.y * 64;
  const unsigned short* X = xb + (size_t)b * SS * CC;

  v8f acc[4];
#pragma unroll
  for (int j = 0; j < 4; ++j)
#pragma unroll
    for (int r = 0; r < 8; ++r) acc[j][r] = 0.0f;

  if (s0 >= 2 && s0 + 64 + 2 <= SS)        // scalar (uniform) branch
    conv1_loop<false>(X, wdp, m0, s0, lane, acc);
  else
    conv1_loop<true>(X, wdp, m0, s0, lane, acc);

  const int mb = m0 + 8 * hf;
  float bs[8];
#pragma unroll
  for (int r = 0; r < 8; ++r) bs[r] = bias[mb + r];
#pragma unroll
  for (int j = 0; j < 4; ++j) {
    int s = s0 + j * 16 + nl;
    float* yp = y + ((size_t)b * SS + s) * CC + mb;
#pragma unroll
    for (int r = 0; r < 8; ++r) yp[r] = acc[j][r] + bs[r];
  }
}

// ---------------------------------------------------------------------------
// channel LayerNorm: one block per (b,s); 128 thr x 4 ch. (1st LN cancels.)
// ---------------------------------------------------------------------------
__global__ __launch_bounds__(128)
void k_ln(const float* __restrict__ y, const float* __restrict__ g,
          const float* __restrict__ be, unsigned short* __restrict__ zb) {
  const int tid = threadIdx.x;
  const size_t base = (size_t)blockIdx.x * CC;
  float4 v = *(const float4*)(y + base + tid * 4);
  float s = v.x + v.y + v.z + v.w;
  float q = v.x * v.x + v.y * v.y + v.z * v.z + v.w * v.w;
  __shared__ float rs[128], rq[128];
  rs[tid] = s; rq[tid] = q;
  __syncthreads();
  for (int o = 64; o > 0; o >>= 1) {
    if (tid < o) { rs[tid] += rs[tid + o]; rq[tid] += rq[tid + o]; }
    __syncthreads();
  }
  float mu  = rs[0] * (1.0f / CC);
  float var = rq[0] * (1.0f / CC) - mu * mu;
  float rsg = rsqrtf(var + 1e-5f);
  float4 gg = *(const float4*)(g + tid * 4);
  float4 bb = *(const float4*)(be + tid * 4);
  float o0 = (v.x - mu) * rsg * gg.x + bb.x;
  float o1 = (v.y - mu) * rsg * gg.y + bb.y;
  float o2 = (v.z - mu) * rsg * gg.z + bb.z;
  float o3 = (v.w - mu) * rsg * gg.w + bb.w;
  uint2 st;
  st.x = (unsigned)f2bf(o0) | ((unsigned)f2bf(o1) << 16);
  st.y = (unsigned)f2bf(o2) | ((unsigned)f2bf(o3) << 16);
  *(uint2*)(zb + base + tid * 4) = st;
}

// ---------------------------------------------------------------------------
// pw1: h = swish(W1 z + b1), C -> 2C.  grid (2C/128, S/64, B). No padding.
// ---------------------------------------------------------------------------
__global__ __launch_bounds__(256)
void k_pw1(const unsigned short* __restrict__ zb, const unsigned short* __restrict__ w1p,
           const float* __restrict__ b1, unsigned short* __restrict__ hb) {
  const int lane = threadIdx.x & 31, wv = threadIdx.x >> 5;
  const int nl = lane & 15, hf = lane >> 4;
  const int b  = blockIdx.z;
  const int m0 = blockIdx.x * 128 + wv * 16;
  const int s0 = blockIdx.y * 64;
  const unsigned short* X = zb + (size_t)b * SS * CC;

  v8f acc[4];
#pragma unroll
  for (int j = 0; j < 4; ++j)
#pragma unroll
    for (int r = 0; r < 8; ++r) acc[j][r] = 0.0f;

  const unsigned short* A = w1p + (size_t)m0 * CC;
  for (int k = 0; k < CC; k += 32) {
    v16bf af = load_a(A, CC, k, lane);
#pragma unroll
    for (int j = 0; j < 4; ++j) {
      v16bf bf = load_b_fast(X, CC, s0 + j * 16 + nl, k, lane);
      acc[j] = __builtin_amdgcn_wmma_f32_16x16x32_bf16(
          false, af, false, bf, (short)0, acc[j], false, false);
    }
  }

  const int mb = m0 + 8 * hf;
  float bs[8];
#pragma unroll
  for (int r = 0; r < 8; ++r) bs[r] = b1[mb + r];
#pragma unroll
  for (int j = 0; j < 4; ++j) {
    int s = s0 + j * 16 + nl;
    float hv[8];
#pragma unroll
    for (int r = 0; r < 8; ++r) {
      float h = acc[j][r] + bs[r];
      hv[r] = h * sigmoidf(h);                       // Swish
    }
    *(uint4*)(hb + ((size_t)b * SS + s) * C2 + mb) = pack8bf(hv);
  }
}

// ---------------------------------------------------------------------------
// conv2 main loop (templated on EDGE); both GLU halves per wave.
// ---------------------------------------------------------------------------
template <bool EDGE>
__device__ __forceinline__ void conv2_loop(const unsigned short* __restrict__ X,
                                           const unsigned short* __restrict__ w2p,
                                           int m0, int s0, int lane,
                                           v8f aA[4], v8f aG[4]) {
  const int nl = lane & 15;
  for (int t = 0; t < KW; ++t) {
    const unsigned short* Aa = w2p + ((size_t)t * C2 + m0) * C2;
    const unsigned short* Ag = w2p + ((size_t)t * C2 + m0 + CC) * C2;
    for (int k = 0; k < C2; k += 32) {
      v16bf afa = load_a(Aa, C2, k, lane);
      v16bf afg = load_a(Ag, C2, k, lane);
#pragma unroll
      for (int j = 0; j < 4; ++j) {
        int s = s0 + j * 16 + nl + t - 2;
        v16bf bf = load_b<EDGE>(X, C2, s, SS, k, lane);
        aA[j] = __builtin_amdgcn_wmma_f32_16x16x32_bf16(
            false, afa, false, bf, (short)0, aA[j], false, false);
        aG[j] = __builtin_amdgcn_wmma_f32_16x16x32_bf16(
            false, afg, false, bf, (short)0, aG[j], false, false);
      }
    }
  }
}

// conv2 + GLU + BN-stat accumulation.  grid (C/128, S/64, B)
__global__ __launch_bounds__(256)
void k_conv2(const unsigned short* __restrict__ hb, const unsigned short* __restrict__ w2p,
             const float* __restrict__ b2, unsigned short* __restrict__ ub,
             float* __restrict__ ssum, float* __restrict__ ssq) {
  const int lane = threadIdx.x & 31, wv = threadIdx.x >> 5;
  const int nl = lane & 15, hf = lane >> 4;
  const int b  = blockIdx.z;
  const int m0 = blockIdx.x * 128 + wv * 16;        // a-half rows; g-half = +512
  const int s0 = blockIdx.y * 64;
  const unsigned short* X = hb + (size_t)b * SS * C2;

  v8f aA[4], aG[4];
#pragma unroll
  for (int j = 0; j < 4; ++j)
#pragma unroll
    for (int r = 0; r < 8; ++r) { aA[j][r] = 0.0f; aG[j][r] = 0.0f; }

  if (s0 >= 2 && s0 + 64 + 2 <= SS)        // scalar (uniform) branch
    conv2_loop<false>(X, w2p, m0, s0, lane, aA, aG);
  else
    conv2_loop<true>(X, w2p, m0, s0, lane, aA, aG);

  const int mb = m0 + 8 * hf;
  float ba[8], bg[8], lsum[8], lsq[8];
#pragma unroll
  for (int r = 0; r < 8; ++r) {
    ba[r] = b2[mb + r]; bg[r] = b2[mb + CC + r];
    lsum[r] = 0.0f; lsq[r] = 0.0f;
  }
#pragma unroll
  for (int j = 0; j < 4; ++j) {
    int s = s0 + j * 16 + nl;
    float uv[8];
#pragma unroll
    for (int r = 0; r < 8; ++r) {
      float av = aA[j][r] + ba[r];
      float gv = aG[j][r] + bg[r];
      float u  = av * sigmoidf(gv);                  // GLU
      uv[r] = u;
      lsum[r] += u; lsq[r] += u * u;
    }
    *(uint4*)(ub + ((size_t)b * SS + s) * CC + mb) = pack8bf(uv);
  }
  // reduce over the 16 lanes of each half-wave (same m per reg, different n)
#pragma unroll
  for (int off = 1; off < 16; off <<= 1) {
#pragma unroll
    for (int r = 0; r < 8; ++r) {
      lsum[r] += __shfl_xor(lsum[r], off, 16);
      lsq[r]  += __shfl_xor(lsq[r],  off, 16);
    }
  }
  if (nl == 0) {
#pragma unroll
    for (int r = 0; r < 8; ++r) {
      atomicAdd(&ssum[mb + r], lsum[r]);
      atomicAdd(&ssq[mb + r],  lsq[r]);
    }
  }
}

// ---------------------------------------------------------------------------
// BN finalize: per-channel scale/shift.   1 block x 512
// ---------------------------------------------------------------------------
__global__ void k_bnfin(const float* __restrict__ ssum, const float* __restrict__ ssq,
                        const float* __restrict__ bng, const float* __restrict__ bnb,
                        float* __restrict__ scale, float* __restrict__ shift) {
  int c = threadIdx.x;
  const float invN = 1.0f / ((float)BB * SS);
  float mu  = ssum[c] * invN;
  float var = ssq[c] * invN - mu * mu;
  float sc  = bng[c] * rsqrtf(var + 1e-5f);
  scale[c] = sc;
  shift[c] = bnb[c] - mu * sc;
}

// ---------------------------------------------------------------------------
// fold BN into w3:  w3eff = w3*scale (bf16),  b3eff = b3 + w3 . shift
// ---------------------------------------------------------------------------
__global__ __launch_bounds__(128)
void k_foldw3(const float* __restrict__ w3, const float* __restrict__ b3,
              const float* __restrict__ scale, const float* __restrict__ shift,
              unsigned short* __restrict__ w3e, float* __restrict__ b3e) {
  const int co = blockIdx.x, tid = threadIdx.x;
  float acc = 0.0f;
  for (int c = tid; c < CC; c += 128) {
    float w = w3[(size_t)co * CC + c];
    acc += w * shift[c];
    w3e[(size_t)co * CC + c] = f2bf(w * scale[c]);
  }
  __shared__ float red[128];
  red[tid] = acc;
  __syncthreads();
  for (int o = 64; o > 0; o >>= 1) {
    if (tid < o) red[tid] += red[tid + o];
    __syncthreads();
  }
  if (tid == 0) b3e[co] = b3[co] + red[0];
}

// ---------------------------------------------------------------------------
// pw3: out = W3eff u + b3eff  -> f32 (B,S,C) final output. No padding.
// ---------------------------------------------------------------------------
__global__ __launch_bounds__(256)
void k_pw3(const unsigned short* __restrict__ ub, const unsigned short* __restrict__ w3e,
           const float* __restrict__ b3e, float* __restrict__ out) {
  const int lane = threadIdx.x & 31, wv = threadIdx.x >> 5;
  const int nl = lane & 15, hf = lane >> 4;
  const int b  = blockIdx.z;
  const int m0 = blockIdx.x * 128 + wv * 16;
  const int s0 = blockIdx.y * 64;
  const unsigned short* X = ub + (size_t)b * SS * CC;

  v8f acc[4];
#pragma unroll
  for (int j = 0; j < 4; ++j)
#pragma unroll
    for (int r = 0; r < 8; ++r) acc[j][r] = 0.0f;

  const unsigned short* A = w3e + (size_t)m0 * CC;
  for (int k = 0; k < CC; k += 32) {
    v16bf af = load_a(A, CC, k, lane);
#pragma unroll
    for (int j = 0; j < 4; ++j) {
      v16bf bf = load_b_fast(X, CC, s0 + j * 16 + nl, k, lane);
      acc[j] = __builtin_amdgcn_wmma_f32_16x16x32_bf16(
          false, af, false, bf, (short)0, acc[j], false, false);
    }
  }

  const int mb = m0 + 8 * hf;
  float bs[8];
#pragma unroll
  for (int r = 0; r < 8; ++r) bs[r] = b3e[mb + r];
#pragma unroll
  for (int j = 0; j < 4; ++j) {
    int s = s0 + j * 16 + nl;
    float* op = out + ((size_t)b * SS + s) * CC + mb;
#pragma unroll
    for (int r = 0; r < 8; ++r) op[r] = acc[j][r] + bs[r];
  }
}

// ---------------------------------------------------------------------------
// workspace layout (bytes).  xb region reused for u; y region reused for h.
// ---------------------------------------------------------------------------
static constexpr size_t NXE    = (size_t)BB * SS * CC;          // 16.7M elems
static constexpr size_t OFF_XB = 0;                             // bf16 x / later bf16 u
static constexpr size_t OFF_Y  = OFF_XB + NXE * 2;              // f32 y / later bf16 h(2C)
static constexpr size_t OFF_ZB = OFF_Y + (size_t)BB * SS * C2 * 2;
static constexpr size_t OFF_WD = OFF_ZB + NXE * 2;
static constexpr size_t OFF_W1 = OFF_WD + (size_t)KW * CC * CC * 2;
static constexpr size_t OFF_W2 = OFF_W1 + (size_t)C2 * CC * 2;
static constexpr size_t OFF_W3 = OFF_W2 + (size_t)KW * C2 * C2 * 2;
static constexpr size_t OFF_ST = OFF_W3 + (size_t)CC * CC * 2;
static constexpr size_t WS_NEEDED = OFF_ST + 2560 * sizeof(float);

extern "C" void kernel_launch(void* const* d_in, const int* in_sizes, int n_in,
                              void* d_out, int out_size, void* d_ws, size_t ws_size,
                              hipStream_t stream) {
  (void)in_sizes; (void)n_in; (void)out_size;
  if (ws_size < WS_NEEDED) return;

  const float* x    = (const float*)d_in[0];
  const float* wd   = (const float*)d_in[1];
  const float* dcb  = (const float*)d_in[2];
  const float* lng  = (const float*)d_in[3];
  const float* lnb  = (const float*)d_in[4];
  const float* w1   = (const float*)d_in[5];
  const float* b1   = (const float*)d_in[6];
  const float* w2   = (const float*)d_in[7];
  const float* b2   = (const float*)d_in[8];
  const float* bng  = (const float*)d_in[9];
  const float* bnb  = (const float*)d_in[10];
  const float* w3   = (const float*)d_in[11];
  const float* b3   = (const float*)d_in[12];
  float* out = (float*)d_out;

  char* ws = (char*)d_ws;
  unsigned short* xb  = (unsigned short*)(ws + OFF_XB);
  unsigned short* ub  = (unsigned short*)(ws + OFF_XB);   // reuse after conv1
  float*          y   = (float*)(ws + OFF_Y);
  unsigned short* hb  = (unsigned short*)(ws + OFF_Y);    // reuse after LN
  unsigned short* zb  = (unsigned short*)(ws + OFF_ZB);
  unsigned short* wdp = (unsigned short*)(ws + OFF_WD);
  unsigned short* w1p = (unsigned short*)(ws + OFF_W1);
  unsigned short* w2p = (unsigned short*)(ws + OFF_W2);
  unsigned short* w3e = (unsigned short*)(ws + OFF_W3);
  float* st    = (float*)(ws + OFF_ST);
  float* ssum  = st;
  float* ssq   = st + 512;
  float* scale = st + 1024;
  float* shift = st + 1536;
  float* b3e   = st + 2048;

  // 1) casts/packs + zero BN stats
  k_prep<<<4096, 256, 0, stream>>>(x, wd, w1, w2, xb, wdp, w1p, w2p, st);
  // 2) conv1 (WMMA) -> y f32        (1st LayerNorm cancels algebraically)
  k_conv1<<<dim3(CC / 128, SS / 64, BB), 256, 0, stream>>>(xb, wdp, dcb, y);
  // 3) channel LayerNorm -> zb bf16
  k_ln<<<BB * SS, 128, 0, stream>>>(y, lng, lnb, zb);
  // 4) pointwise C->2C + Swish (WMMA) -> hb bf16
  k_pw1<<<dim3(C2 / 128, SS / 64, BB), 256, 0, stream>>>(zb, w1p, b1, hb);
  // 5) conv2 (WMMA, both GLU halves per wave) + GLU + BN stats -> ub bf16
  k_conv2<<<dim3(CC / 128, SS / 64, BB), 256, 0, stream>>>(hb, w2p, b2, ub, ssum, ssq);
  // 6) BN finalize + fold into w3
  k_bnfin<<<1, 512, 0, stream>>>(ssum, ssq, bng, bnb, scale, shift);
  k_foldw3<<<512, 128, 0, stream>>>(w3, b3, scale, shift, w3e, b3e);
  // 7) pointwise C->C (WMMA) -> out f32 (B,S,C)
  k_pw3<<<dim3(CC / 128, SS / 64, BB), 256, 0, stream>>>(ub, w3e, b3e, out);
}
